// BornFNOV3_25812753449220
// MI455X (gfx1250) — matrix operand order repaired and checked by reading, not compiled, not run on hardware
//
#include <hip/hip_runtime.h>
#include <math.h>

typedef __bf16 bf16;
typedef __attribute__((ext_vector_type(16))) __bf16 v16bf;
typedef __attribute__((ext_vector_type(8)))  __bf16 v8bf;
typedef __attribute__((ext_vector_type(8)))  float   v8f;

typedef unsigned int u32x4 __attribute__((ext_vector_type(4)));
typedef int          i32x8 __attribute__((ext_vector_type(8)));
typedef int          i32x4 __attribute__((ext_vector_type(4)));

#if defined(__has_builtin)
#if __has_builtin(__builtin_amdgcn_tensor_load_to_lds) && __has_builtin(__builtin_amdgcn_s_wait_tensorcnt)
#define USE_TDM 1
#endif
#endif
#ifndef USE_TDM
#define USE_TDM 0
#endif

#define B_    2
#define H_    480
#define W_    480
#define C_    32
#define HALF_ 16
#define L_    4
#define WN_   20
#define HP_   500        // padded (reference) spatial size
#define SDIM  512        // GEMM-padded spatial size
#define SPAT  (SDIM*SDIM)
#define KX_   40         // kept row-modes: 0..19 and 480..499
#define MROW  96         // stacked Re(48) + Im(48) rows (40 valid each)
#define NIMG  (B_*C_)

__device__ __forceinline__ float gelu_f(float x) {
    return 0.5f * x * (1.0f + erff(x * 0.70710678118654752f));
}
__device__ __forceinline__ bf16 tobf(float x) { return (bf16)x; }

// ---------------------------------------------------------------------------
// Build bf16 DFT bases and bf16 conv weights.
//   Fbig[96][512]: rows 0..39  =  cos(2*pi*kx*h/500)   (forward Re)
//                  rows 48..87 = -sin(2*pi*kx*h/500)   (forward Im)
//   A2[512][96]:   cols 0..39  =  cos(2*pi*kx*h/500)/N^2
//                  cols 48..87 = -sin(2*pi*kx*h/500)/N^2   (inverse, Re output)
// ---------------------------------------------------------------------------
__global__ void build_basis(bf16* __restrict__ Fbig, bf16* __restrict__ A2,
                            bf16* __restrict__ cwbf, const float* __restrict__ conv_w)
{
    int idx = blockIdx.x * blockDim.x + threadIdx.x;
    const float TWO_PI = 6.2831853071795864769f;
    const int nF = MROW * SDIM;
    const int nA = SDIM * MROW;
    if (idx < nF) {
        int m = idx / SDIM, k = idx % SDIM;
        float v = 0.f;
        if (k < HP_) {
            if (m < KX_) {
                int kx = (m < WN_) ? m : (480 + (m - WN_));
                int ph = (int)(((long long)kx * k) % HP_);
                v = cosf(TWO_PI * (float)ph / (float)HP_);
            } else if (m >= 48 && m < 48 + KX_) {
                int mm = m - 48;
                int kx = (mm < WN_) ? mm : (480 + (mm - WN_));
                int ph = (int)(((long long)kx * k) % HP_);
                v = -sinf(TWO_PI * (float)ph / (float)HP_);
            }
        }
        Fbig[idx] = tobf(v);
    } else if (idx < nF + nA) {
        int j = idx - nF;
        int h = j / MROW, k = j % MROW;
        const float invN2 = 1.0f / (500.0f * 500.0f);
        float v = 0.f;
        if (k < KX_) {
            int kx = (k < WN_) ? k : (480 + (k - WN_));
            int ph = (int)(((long long)kx * h) % HP_);
            v = cosf(TWO_PI * (float)ph / (float)HP_) * invN2;
        } else if (k >= 48 && k < 48 + KX_) {
            int kk = k - 48;
            int kx = (kk < WN_) ? kk : (480 + (kk - WN_));
            int ph = (int)(((long long)kx * h) % HP_);
            v = -sinf(TWO_PI * (float)ph / (float)HP_) * invN2;
        }
        A2[j] = tobf(v);
    } else if (idx < nF + nA + L_ * C_ * C_) {
        int j = idx - nF - nA;
        cwbf[j] = tobf(conv_w[j]);
    }
}

// ---------------------------------------------------------------------------
// Encoder: two small MLPs per pixel -> a, e laid out [b][c][512*512] (f32),
// zero outside the valid 480x480 region (reference PAD band and GEMM pad).
// ---------------------------------------------------------------------------
__global__ void encode_kernel(const float* __restrict__ input_data,
                              const float* __restrict__ src_data,
                              const float* __restrict__ in_w1, const float* __restrict__ in_b1,
                              const float* __restrict__ in_w2, const float* __restrict__ in_b2,
                              const float* __restrict__ ep_w1, const float* __restrict__ ep_b1,
                              const float* __restrict__ ep_w2, const float* __restrict__ ep_b2,
                              float* __restrict__ a, float* __restrict__ e)
{
    int idx = blockIdx.x * blockDim.x + threadIdx.x;
    if (idx >= B_ * SPAT) return;
    int b = idx / SPAT, p = idx % SPAT;
    int h = p >> 9, w = p & (SDIM - 1);
    float av[C_], ev[C_];
    if (h < H_ && w < W_) {
        float gx = (float)h * (1.0f / 479.0f);
        float gy = (float)w * (1.0f / 479.0f);
        const float* sp = src_data + ((long long)(b * H_ + h) * W_ + w) * 3;
        float xin[5] = { sp[0], sp[1], sp[2], gx, gy };
        float xep[3] = { input_data[(long long)(b * H_ + h) * W_ + w], gx, gy };
        float h1[HALF_];
        #pragma unroll
        for (int j = 0; j < HALF_; ++j) {
            float s = in_b1[j];
            #pragma unroll
            for (int f = 0; f < 5; ++f) s += xin[f] * in_w1[f * HALF_ + j];
            h1[j] = gelu_f(s);
        }
        #pragma unroll
        for (int c = 0; c < C_; ++c) {
            float s = in_b2[c];
            #pragma unroll
            for (int j = 0; j < HALF_; ++j) s += h1[j] * in_w2[j * C_ + c];
            av[c] = s;
        }
        float h2[HALF_];
        #pragma unroll
        for (int j = 0; j < HALF_; ++j) {
            float s = ep_b1[j];
            #pragma unroll
            for (int f = 0; f < 3; ++f) s += xep[f] * ep_w1[f * HALF_ + j];
            h2[j] = gelu_f(s);
        }
        #pragma unroll
        for (int c = 0; c < C_; ++c) {
            float s = ep_b2[c];
            #pragma unroll
            for (int j = 0; j < HALF_; ++j) s += h2[j] * ep_w2[j * C_ + c];
            ev[c] = s;
        }
    } else {
        #pragma unroll
        for (int c = 0; c < C_; ++c) { av[c] = 0.f; ev[c] = 0.f; }
    }
    #pragma unroll
    for (int c = 0; c < C_; ++c) {
        long long o = (long long)(b * C_ + c) * SPAT + p;
        a[o] = av[c];
        e[o] = ev[c];
    }
}

// T = bf16(a*e), a_bf = bf16(a)
__global__ void pack_kernel(const float* __restrict__ a, const float* __restrict__ e,
                            bf16* __restrict__ Tb, bf16* __restrict__ ab)
{
    long long idx = (long long)blockIdx.x * blockDim.x + threadIdx.x;
    if (idx >= (long long)NIMG * SPAT) return;
    float x = a[idx];
    Tb[idx] = tobf(x * e[idx]);
    ab[idx] = tobf(x);
}

// ---------------------------------------------------------------------------
// Generic bf16 WMMA GEMM:  C[img] (+)= A x B[img]
//   A: bf16 [M x K] row-major (ldA), shared across imgs
//   B: bf16 [K x N] row-major (ldB) per img
//   C: f32  (ldC) per img
// Requires M%32==0 (grid.y covers M/32), N%64==0 (grid.x), K%32==0.
// 4 waves / block; each wave owns a 16-wide N slice of the 64-wide LDS tile
// and computes a 32x16 output patch (two WMMA tiles sharing the B fragment).
// B tile staging: TDM tensor_load_to_lds (one wave issues the 32x64 bf16
// 2-D descriptor, waits TENSORcnt) when available; cooperative vector
// loads otherwise.
// ---------------------------------------------------------------------------
__global__ __launch_bounds__(128)
void gemm_bf16_wmma(const bf16* __restrict__ A, const bf16* __restrict__ Bm,
                    float* __restrict__ Cm, int K, int ldA, int ldB, int ldC,
                    long long strideBimg, long long strideCimg, int accumulate)
{
    __shared__ bf16 Bs[32 * 64];
    const int img  = blockIdx.z;
    const bf16* Bp = Bm + (long long)img * strideBimg;
    float*      Cp = Cm + (long long)img * strideCimg;
    const int wave = threadIdx.x >> 5;
    const int lane = threadIdx.x & 31;
    const int m0 = blockIdx.y * 32;
    const int nb = blockIdx.x * 64;

    v8f acc0, acc1;
    #pragma unroll
    for (int i = 0; i < 8; ++i) { acc0[i] = 0.f; acc1[i] = 0.f; }

    // ISA 16-bit A layout: lanes 0-15 hold K{0..7,16..23}, lanes 16-31 K{8..15,24..31}
    const int arow = m0 + (lane & 15);
    const int koff = (lane < 16) ? 0 : 8;
    // cooperative-staging fallback indices: 128 threads x 16 bf16 = 32x64 tile
    const int br = threadIdx.x >> 2;
    const int bc = (threadIdx.x & 3) * 16;
    // ISA 16-bit B layout: lane = N column, elements = K {0..7,16..23} / {8..15,24..31}
    const int cl = (wave << 4) + (lane & 15);

#if USE_TDM
    const unsigned lds_base = (unsigned)(unsigned long long)(&Bs[0]);
#endif

    for (int k0 = 0; k0 < K; k0 += 32) {
        __syncthreads();
#if USE_TDM
        if (wave == 0) {
            // Tensor DMA descriptor: 2-D tile, 32 rows (stride ldB) x 64 bf16
            unsigned long long ga =
                (unsigned long long)(const void*)(Bp + (long long)k0 * ldB + nb);
            u32x4 g0;
            g0[0] = 1u;                                   // count=1, user mode
            g0[1] = lds_base;                             // lds_addr (bytes)
            g0[2] = (unsigned)(ga & 0xffffffffu);         // global_addr[31:0]
            g0[3] = (unsigned)((ga >> 32) & 0x01ffffffu)  // global_addr[56:32]
                    | (2u << 30);                         // type=2 ("image")
            i32x8 g1;
            g1[0] = (1 << 16);        // workgroup_mask=0, data_size=1 (2 bytes)
            g1[1] = (64 << 16);       // tensor_dim0[15:0]=64  (abar_addr=0)
            g1[2] = (32 << 16);       // tensor_dim0 hi=0 | tensor_dim1 lo=32
            g1[3] = (64 << 16);       // tensor_dim1 hi=0 | tile_dim0=64
            g1[4] = 32;               // tile_dim1=32, tile_dim2=0
            g1[5] = ldB;              // tensor_dim0_stride[31:0] (elements)
            g1[6] = 0;                // stride hi / dim1_stride lo
            g1[7] = 0;
            i32x4 gz; gz[0] = 0; gz[1] = 0; gz[2] = 0; gz[3] = 0;
            i32x8 gz8;
            #pragma unroll
            for (int i = 0; i < 8; ++i) gz8[i] = 0;
            __builtin_amdgcn_tensor_load_to_lds(g0, g1, gz, gz, gz8, 0);
            __builtin_amdgcn_s_wait_tensorcnt(0);
        }
#else
        {
            const bf16* src = Bp + (long long)(k0 + br) * ldB + nb + bc;
            v8bf b0 = *(const v8bf*)(src);
            v8bf b1 = *(const v8bf*)(src + 8);
            *(v8bf*)(&Bs[br * 64 + bc])     = b0;
            *(v8bf*)(&Bs[br * 64 + bc + 8]) = b1;
        }
#endif
        __syncthreads();

        const bf16* ap = A + (long long)arow * ldA + k0 + koff;
        v8bf a0 = *(const v8bf*)(ap);
        v8bf a1 = *(const v8bf*)(ap + 16);
        const bf16* ap2 = ap + (long long)16 * ldA;
        v8bf a2 = *(const v8bf*)(ap2);
        v8bf a3 = *(const v8bf*)(ap2 + 16);
        if (k0 + 32 < K) __builtin_prefetch((const void*)(ap + 32), 0, 0);

        v16bf afrag0, afrag1;
        #pragma unroll
        for (int j = 0; j < 8; ++j) {
            afrag0[j] = a0[j]; afrag0[8 + j] = a1[j];
            afrag1[j] = a2[j]; afrag1[8 + j] = a3[j];
        }

        v16bf bfrag;
        #pragma unroll
        for (int j = 0; j < 16; ++j) bfrag[j] = Bs[(koff + (j & 7) + ((j >> 3) << 4)) * 64 + cl];

        acc0 = __builtin_amdgcn_wmma_f32_16x16x32_bf16(
            false, afrag0, false, bfrag, (short)0, acc0, false, false);
        acc1 = __builtin_amdgcn_wmma_f32_16x16x32_bf16(
            false, afrag1, false, bfrag, (short)0, acc1, false, false);
    }

    // D layout: lane<16 -> N=lane, M rows +0..7 ; lane>=16 -> M rows +8..15
    const int gc = nb + (wave << 4) + (lane & 15);
    const int mr = m0 + ((lane < 16) ? 0 : 8);
    #pragma unroll
    for (int r = 0; r < 8; ++r) {
        long long o0 = (long long)(mr + r) * ldC + gc;
        long long o1 = (long long)(mr + 16 + r) * ldC + gc;
        float v0 = acc0[r];
        float v1 = acc1[r];
        if (accumulate) { v0 += Cp[o0]; v1 += Cp[o1]; }
        Cp[o0] = v0;
        Cp[o1] = v1;
    }
}

// Column DFT: V[img][r][ky] = sum_w U[img][r][w] * exp(-2*pi*i*ky*w/500)
__global__ void coldft_kernel(const float* __restrict__ U, float* __restrict__ V)
{
    int idx = blockIdx.x * blockDim.x + threadIdx.x;
    if (idx >= NIMG * KX_ * WN_) return;
    int img = idx / (KX_ * WN_);
    int rem = idx % (KX_ * WN_);
    int r = rem / WN_, ky = rem % WN_;
    const float* Ur = U + (long long)img * MROW * SDIM + (long long)r * SDIM;
    const float* Ui = Ur + 48 * SDIM;
    float vr = 0.f, vi = 0.f;
    for (int w = 0; w < HP_; ++w) {
        int ph = (ky * w) % HP_;
        float th = 6.2831853071795864769f * (float)ph / (float)HP_;
        float c = cosf(th), s = sinf(th);
        float ur = Ur[w], ui = Ui[w];
        vr += ur * c + ui * s;
        vi += ui * c - ur * s;
    }
    float* vp = V + (long long)idx * 2;
    vp[0] = vr; vp[1] = vi;
}

// Spectral channel mixing: independent Re/Im scaling (shared z in the einsum)
__global__ void spectral_kernel(const float* __restrict__ V,
                                const float* __restrict__ fw1,
                                const float* __restrict__ fw2,
                                int l, float* __restrict__ S)
{
    int idx = blockIdx.x * blockDim.x + threadIdx.x;
    if (idx >= B_ * C_ * KX_ * WN_) return;
    int y = idx % WN_;
    int t = idx / WN_;
    int x = t % KX_; t /= KX_;
    int o = t % C_;
    int b = t / C_;
    const float* fw = (x < WN_) ? fw1 : fw2;
    int xx = (x < WN_) ? x : (x - WN_);
    float sr = 0.f, si = 0.f;
    for (int i = 0; i < C_; ++i) {
        const float* vp = V + ((((long long)(b * C_ + i) * KX_ + x) * WN_) + y) * 2;
        long long wi = (((((long long)l * C_ + i) * C_ + o) * WN_ + xx) * WN_ + y) * 2;
        sr += vp[0] * fw[wi + 0];
        si += vp[1] * fw[wi + 1];
    }
    float* sp = S + ((((long long)(b * C_ + o) * KX_ + x) * WN_) + y) * 2;
    sp[0] = sr; sp[1] = si;
}

// Width inverse: P[img][r][w] stacked Re/Im, Hermitian doubling (c0=1, ck=2)
__global__ void widthinv_kernel(const float* __restrict__ S, bf16* __restrict__ P)
{
    int idx = blockIdx.x * blockDim.x + threadIdx.x;
    if (idx >= NIMG * MROW * SDIM) return;
    int w = idx % SDIM;
    int t = idx / SDIM;
    int r = t % MROW;
    int img = t / MROW;
    float v = 0.f;
    if (w < HP_) {
        if (r < KX_) {
            const float* sp = S + ((long long)img * KX_ + r) * WN_ * 2;
            for (int ky = 0; ky < WN_; ++ky) {
                float ck = (ky == 0) ? 1.f : 2.f;
                int ph = (ky * w) % HP_;
                float th = 6.2831853071795864769f * (float)ph / (float)HP_;
                v += ck * (sp[ky * 2 + 0] * cosf(th) - sp[ky * 2 + 1] * sinf(th));
            }
        } else if (r >= 48 && r < 48 + KX_) {
            const float* sp = S + ((long long)img * KX_ + (r - 48)) * WN_ * 2;
            for (int ky = 0; ky < WN_; ++ky) {
                float ck = (ky == 0) ? 1.f : 2.f;
                int ph = (ky * w) % HP_;
                float th = 6.2831853071795864769f * (float)ph / (float)HP_;
                v += ck * (sp[ky * 2 + 0] * sinf(th) + sp[ky * 2 + 1] * cosf(th));
            }
        }
    }
    P[idx] = tobf(v);
}

// a = act(fourier + conv + bias) over valid 500x500 region
__global__ void combine_kernel(const float* __restrict__ fout,
                               const float* __restrict__ conv_b,
                               int l, int apply_act, float* __restrict__ a)
{
    long long idx = (long long)blockIdx.x * blockDim.x + threadIdx.x;
    if (idx >= (long long)NIMG * SPAT) return;
    int p = (int)(idx % SPAT);
    int h = p >> 9, w = p & (SDIM - 1);
    if (h >= HP_ || w >= HP_) return;   // keep GEMM-pad band at zero
    int c = (int)((idx / SPAT) % C_);
    float v = fout[idx] + conv_b[l * C_ + c];
    if (apply_act) v = gelu_f(v);
    a[idx] = v;
}

// Projection MLP + 3x3 mask conv + field blend
__global__ void final_kernel(const float* __restrict__ a,
                             const float* __restrict__ input_data,
                             const float* __restrict__ src_data,
                             const float* __restrict__ pw1, const float* __restrict__ pb1,
                             const float* __restrict__ pw2, const float* __restrict__ pb2,
                             const float* __restrict__ mw,  const float* __restrict__ mb,
                             float* __restrict__ out)
{
    int idx = blockIdx.x * blockDim.x + threadIdx.x;
    if (idx >= B_ * H_ * W_) return;
    int w = idx % W_;
    int t = idx / W_;
    int h = t % H_;
    int b = t / H_;

    float av[C_];
    #pragma unroll
    for (int c = 0; c < C_; ++c)
        av[c] = a[(long long)(b * C_ + c) * SPAT + (h << 9) + w];

    float h1[HALF_];
    #pragma unroll
    for (int j = 0; j < HALF_; ++j) {
        float s = pb1[j];
        #pragma unroll
        for (int c = 0; c < C_; ++c) s += av[c] * pw1[c * HALF_ + j];
        h1[j] = gelu_f(s);
    }
    float p0 = pb2[0], p1 = pb2[1];
    #pragma unroll
    for (int j = 0; j < HALF_; ++j) { p0 += h1[j] * pw2[j * 2 + 0]; p1 += h1[j] * pw2[j * 2 + 1]; }

    float m = mb[0];
    #pragma unroll
    for (int ky = 0; ky < 3; ++ky)
        #pragma unroll
        for (int kx = 0; kx < 3; ++kx) {
            int hh = h + ky - 1, ww = w + kx - 1;
            if (hh >= 0 && hh < H_ && ww >= 0 && ww < W_)
                m += input_data[(long long)(b * H_ + hh) * W_ + ww] * mw[ky * 3 + kx];
        }

    const float* sp = src_data + ((long long)(b * H_ + h) * W_ + w) * 3;
    float* op = out + ((long long)(b * H_ + h) * W_ + w) * 2;
    op[0] = sp[1] * m + p0;
    op[1] = sp[2] * m + p1;
}

// ---------------------------------------------------------------------------
extern "C" void kernel_launch(void* const* d_in, const int* in_sizes, int n_in,
                              void* d_out, int out_size, void* d_ws, size_t ws_size,
                              hipStream_t stream)
{
    (void)in_sizes; (void)n_in; (void)out_size; (void)ws_size;
    const float* input_data = (const float*)d_in[0];
    const float* src_data   = (const float*)d_in[1];
    const float* in_w1 = (const float*)d_in[2];
    const float* in_b1 = (const float*)d_in[3];
    const float* in_w2 = (const float*)d_in[4];
    const float* in_b2 = (const float*)d_in[5];
    const float* ep_w1 = (const float*)d_in[6];
    const float* ep_b1 = (const float*)d_in[7];
    const float* ep_w2 = (const float*)d_in[8];
    const float* ep_b2 = (const float*)d_in[9];
    const float* fw1    = (const float*)d_in[10];
    const float* fw2    = (const float*)d_in[11];
    const float* conv_w = (const float*)d_in[12];
    const float* conv_b = (const float*)d_in[13];
    const float* pw1 = (const float*)d_in[14];
    const float* pb1 = (const float*)d_in[15];
    const float* pw2 = (const float*)d_in[16];
    const float* pb2 = (const float*)d_in[17];
    const float* mw  = (const float*)d_in[18];
    const float* mb  = (const float*)d_in[19];
    float* out = (float*)d_out;

    // workspace carve-up (256B-aligned chunks)
    char* wsb = (char*)d_ws;
    size_t off = 0;
    auto carve = [&](size_t bytes) -> void* {
        void* p = (void*)(wsb + off);
        off += (bytes + 255) & ~(size_t)255;
        return p;
    };
    float* a    = (float*)carve((size_t)NIMG * SPAT * 4);
    float* e    = (float*)carve((size_t)NIMG * SPAT * 4);
    bf16*  ab   = (bf16*) carve((size_t)NIMG * SPAT * 2);
    bf16*  Tb   = (bf16*) carve((size_t)NIMG * SPAT * 2);
    float* U    = (float*)carve((size_t)NIMG * MROW * SDIM * 4);
    float* V    = (float*)carve((size_t)NIMG * KX_ * WN_ * 2 * 4);
    float* S    = (float*)carve((size_t)NIMG * KX_ * WN_ * 2 * 4);
    bf16*  P    = (bf16*) carve((size_t)NIMG * MROW * SDIM * 2);
    float* fout = (float*)carve((size_t)NIMG * SPAT * 4);
    bf16*  Fbig = (bf16*) carve((size_t)MROW * SDIM * 2);
    bf16*  A2   = (bf16*) carve((size_t)SDIM * MROW * 2);
    bf16*  cwbf = (bf16*) carve((size_t)L_ * C_ * C_ * 2);

    const int TPB = 256;
    const int nBasis = MROW * SDIM + SDIM * MROW + L_ * C_ * C_;

    build_basis<<<(nBasis + TPB - 1) / TPB, TPB, 0, stream>>>(Fbig, A2, cwbf, conv_w);

    encode_kernel<<<(B_ * SPAT + TPB - 1) / TPB, TPB, 0, stream>>>(
        input_data, src_data, in_w1, in_b1, in_w2, in_b2,
        ep_w1, ep_b1, ep_w2, ep_b2, a, e);

    const long long strT  = (long long)SPAT;          // T per img
    const long long strU  = (long long)MROW * SDIM;   // U / P per img
    const long long strF  = (long long)SPAT;          // fout per img

    for (int l = 0; l < L_; ++l) {
        pack_kernel<<<(int)(((long long)NIMG * SPAT + TPB - 1) / TPB), TPB, 0, stream>>>(a, e, Tb, ab);

        // forward row-DFT: U(96x512) = Fbig(96x512) x T(512x512), 64 imgs
        gemm_bf16_wmma<<<dim3(SDIM / 64, MROW / 32, NIMG), 128, 0, stream>>>(
            Fbig, Tb, U, /*K=*/SDIM, /*ldA=*/SDIM, /*ldB=*/SDIM, /*ldC=*/SDIM,
            strT, strU, /*acc=*/0);

        coldft_kernel<<<(NIMG * KX_ * WN_ + TPB - 1) / TPB, TPB, 0, stream>>>(U, V);

        spectral_kernel<<<(B_ * C_ * KX_ * WN_ + TPB - 1) / TPB, TPB, 0, stream>>>(V, fw1, fw2, l, S);

        widthinv_kernel<<<(NIMG * MROW * SDIM + TPB - 1) / TPB, TPB, 0, stream>>>(S, P);

        // inverse height: fout(512x512) = A2(512x96) x P(96x512), 64 imgs
        gemm_bf16_wmma<<<dim3(SDIM / 64, SDIM / 32, NIMG), 128, 0, stream>>>(
            A2, P, fout, /*K=*/MROW, /*ldA=*/MROW, /*ldB=*/SDIM, /*ldC=*/SDIM,
            strU, strF, /*acc=*/0);

        // 1x1 conv accumulated into fout: (32x32) x (32 x 512^2), per batch
        gemm_bf16_wmma<<<dim3(SPAT / 64, C_ / 32, B_), 128, 0, stream>>>(
            cwbf + (size_t)l * C_ * C_, ab, fout, /*K=*/C_, /*ldA=*/C_,
            /*ldB=*/SPAT, /*ldC=*/SPAT,
            (long long)C_ * SPAT, (long long)C_ * SPAT, /*acc=*/1);

        combine_kernel<<<(int)(((long long)NIMG * SPAT + TPB - 1) / TPB), TPB, 0, stream>>>(
            fout, conv_b, l, (l < L_ - 1) ? 1 : 0, a);
    }

    final_kernel<<<(B_ * H_ * W_ + TPB - 1) / TPB, TPB, 0, stream>>>(
        a, input_data, src_data, pw1, pb1, pw2, pb2, mw, mb, out);
}
